// MLP_sparse_deep2_54752243090113
// MI455X (gfx1250) — matrix operand order repaired and checked by reading, not compiled, not run on hardware
//
#include <hip/hip_runtime.h>

typedef _Float16 v16h __attribute__((ext_vector_type(16)));
typedef _Float16 v8h  __attribute__((ext_vector_type(8)));
typedef float    v8f  __attribute__((ext_vector_type(8)));
typedef unsigned int u32x4 __attribute__((ext_vector_type(4)));
typedef int          i32x4 __attribute__((ext_vector_type(4)));
typedef int          i32x8 __attribute__((ext_vector_type(8)));

#define LDSW 40   // LDS row stride in halves (32 data + 8 pad -> conflict-free)

#if __has_builtin(__builtin_amdgcn_tensor_load_to_lds) && \
    __has_builtin(__builtin_amdgcn_s_wait_tensorcnt)
#define TDM_OK 1
#else
#define TDM_OK 0
#endif

#if TDM_OK
// Issue one TDM descriptor: load a 128 x 32(f16) tile, global row stride Kp
// elements, into LDS with 16 pad bytes after every 64-byte row (-> LDSW=40).
__device__ __forceinline__ void tdm_load_b_tile(const _Float16* gsrc,
                                                unsigned lds_off, int Kp)
{
    unsigned long long ga = (unsigned long long)(size_t)gsrc;
    u32x4 g0;
    g0[0] = 1u;                                            // count=1, user mode
    g0[1] = lds_off;                                       // LDS byte address
    g0[2] = (unsigned)(ga & 0xFFFFFFFFu);                  // global addr [95:64]
    g0[3] = (unsigned)((ga >> 32) & 0x01FFFFFFu)           // global addr [120:96]
            | (2u << 30);                                  // type=2 ("image")
    i32x8 g1;
    g1[0] = (1 << 16)      // data_size = 1 -> 2 bytes
          | (1 << 20)      // pad_enable
          | (3 << 22)      // pad_interval: code 3 -> every 16 DWORDs (64B row)
          | (3 << 25);     // pad_amount:   code 3 -> 4 DWORDs (16B)
    g1[1] = (int)(((unsigned)Kp & 0xFFFFu) << 16);         // tensor_dim0 lo16
    g1[2] = (int)((((unsigned)Kp >> 16) & 0xFFFFu)         // tensor_dim0 hi16
            | (128u << 16));                               // tensor_dim1 lo16
    g1[3] = (int)(32u << 16);                              // tile_dim0 = 32
    g1[4] = 128;                                           // tile_dim1 = 128
    g1[5] = Kp;                                            // dim0_stride lo32
    g1[6] = 0;
    g1[7] = 0;
    i32x4 z4 = {0, 0, 0, 0};
#if defined(__clang_major__) && (__clang_major__ >= 23)
    i32x8 z8 = {0, 0, 0, 0, 0, 0, 0, 0};
    __builtin_amdgcn_tensor_load_to_lds(g0, g1, z4, z4, z8, 0);
#else
    __builtin_amdgcn_tensor_load_to_lds(g0, g1, z4, z4, 0);
#endif
}
#endif

// ---------------------------------------------------------------------------
// Prep: Wp[Np,Kp] (f16, zero padded) = W * mask ; bp[Np] = padded bias
// ---------------------------------------------------------------------------
__global__ __launch_bounds__(256) void prep_weights(
    const float* __restrict__ W, const unsigned char* __restrict__ mask,
    const float* __restrict__ b, _Float16* __restrict__ Wp,
    float* __restrict__ bp, int N, int K, int Np, int Kp)
{
    long total = (long)Np * Kp;
    for (long idx = (long)blockIdx.x * 256 + threadIdx.x; idx < total;
         idx += (long)gridDim.x * 256) {
        int n = (int)(idx / Kp);
        int k = (int)(idx % Kp);
        float v = 0.0f;
        if (n < N && k < K) {
            long s = (long)n * K + k;
            if (mask[s]) v = W[s];
        }
        Wp[idx] = (_Float16)v;
        if (idx < Np) bp[idx] = (idx < N) ? b[idx] : 0.0f;
    }
}

// ---------------------------------------------------------------------------
// WMMA GEMM: out[M,Nsrc] = relu?( A[M,Ksrc] @ Wp[Np,Kp]^T + bias )
// Block tile 128x128, 8 waves, wave tile 32(M) x 64(N) = 2x4 WMMA 16x16x32 f16
// ---------------------------------------------------------------------------
__global__ __launch_bounds__(256) void gemm_wmma_f16(
    const float* __restrict__ A, int Ksrc, int Kp,
    const _Float16* __restrict__ Bw, const float* __restrict__ bias,
    float* __restrict__ out, float* __restrict__ out2,
    int Nsrc, int do_relu)
{
    __shared__ _Float16 lA[128 * LDSW];
    __shared__ _Float16 lB[128 * LDSW];

    const int tid  = threadIdx.x;
    const int lane = tid & 31;
    const int wave = tid >> 5;
    const int wm   = wave & 3;     // M offset 32*wm within block
    const int wn   = wave >> 2;    // N offset 64*wn within block
    const long mblk = (long)blockIdx.x * 128;
    const long nblk = (long)blockIdx.y * 128;

    // staging: thread covers row (tid>>1), 16 cols at 16*(tid&1)
    const int srow = tid >> 1;
    const int scol = (tid & 1) * 16;

    // per-lane fragment geometry (ISA 7.12.2)
    const int rl = lane & 15;            // row within 16x16 frag
    const int kA = 8  * (lane >> 4);     // A: K base (interleaved halves)
    const int kB = 8  * (lane >> 4);     // B: K base (interleaved halves)
    const int mh = 8  * (lane >> 4);     // C/D: M offset of upper half-wave

#if TDM_OK
    const unsigned ldsB_off = (unsigned)(size_t)(&lB[0]);
#endif

    v8f acc[2][4];
#pragma unroll
    for (int mi = 0; mi < 2; ++mi)
#pragma unroll
        for (int ni = 0; ni < 4; ++ni) acc[mi][ni] = v8f{};

    for (int kt = 0; kt < Kp; kt += 32) {
        __syncthreads();   // previous tile fully consumed

        // ---- stage B tile (padded f16 weights, always in range) ----
#if TDM_OK
        if (tid < 32)
            tdm_load_b_tile(Bw + nblk * (long)Kp + kt, ldsB_off, Kp);
#else
        {
            const _Float16* wp = Bw + (nblk + srow) * (long)Kp + kt + scol;
            v8h b0 = *(const v8h*)wp;
            v8h b1 = *(const v8h*)(wp + 8);
            *(v8h*)&lB[srow * LDSW + scol]     = b0;
            *(v8h*)&lB[srow * LDSW + scol + 8] = b1;
        }
#endif

        // ---- stage A tile (f32 global -> f16 LDS) ----
        {
            const float* ap = A + (mblk + srow) * (long)Ksrc + kt + scol;
            _Float16 tmp[16];
            if (kt + 32 <= Ksrc) {            // block-uniform: no exec masking
#pragma unroll
                for (int i = 0; i < 16; ++i) tmp[i] = (_Float16)ap[i];
                if (kt + 32 < Kp) __builtin_prefetch(ap + 32, 0, 3);
            } else {                          // at most one tail iteration
#pragma unroll
                for (int i = 0; i < 16; ++i) {
                    int k = kt + scol + i;
                    tmp[i] = (_Float16)((k < Ksrc) ? ap[i] : 0.0f);
                }
            }
            *(v8h*)&lA[srow * LDSW + scol]     = *(v8h*)&tmp[0];
            *(v8h*)&lA[srow * LDSW + scol + 8] = *(v8h*)&tmp[8];
        }

#if TDM_OK
        if (tid < 32) __builtin_amdgcn_s_wait_tensorcnt(0);
#endif
        __syncthreads();   // tile visible to all waves

        // ---- gather fragments ----
        v16h af[2], bf[4];
#pragma unroll
        for (int mi = 0; mi < 2; ++mi) {
            int row = 32 * wm + 16 * mi + rl;
            union { v16h v; v8h h[2]; } u;
            u.h[0] = *(const v8h*)&lA[row * LDSW + kA];        // K kA..kA+7
            u.h[1] = *(const v8h*)&lA[row * LDSW + kA + 16];   // K kA+16..kA+23
            af[mi] = u.v;
        }
#pragma unroll
        for (int ni = 0; ni < 4; ++ni) {
            int row = 64 * wn + 16 * ni + rl;
            union { v16h v; v8h h[2]; } u;
            u.h[0] = *(const v8h*)&lB[row * LDSW + kB];        // K kB..kB+7
            u.h[1] = *(const v8h*)&lB[row * LDSW + kB + 16];   // K kB+16..kB+23
            bf[ni] = u.v;
        }

        // ---- 8 WMMAs ----
#pragma unroll
        for (int mi = 0; mi < 2; ++mi)
#pragma unroll
            for (int ni = 0; ni < 4; ++ni)
                acc[mi][ni] = __builtin_amdgcn_wmma_f32_16x16x32_f16(
                    false, af[mi], false, bf[ni], (short)0, acc[mi][ni],
                    false, false);
    }

    // ---- epilogue: bias (+relu), write f32 ----
#pragma unroll
    for (int mi = 0; mi < 2; ++mi) {
#pragma unroll
        for (int ni = 0; ni < 4; ++ni) {
            long n = nblk + 64 * wn + 16 * ni + rl;
            float bv = bias[n];
            if (n < Nsrc) {
#pragma unroll
                for (int e = 0; e < 8; ++e) {
                    long m = mblk + 32 * wm + 16 * mi + mh + e;
                    float v = acc[mi][ni][e] + bv;
                    if (do_relu) v = v > 0.0f ? v : 0.0f;
                    out[m * Nsrc + n] = v;
                    if (out2) out2[m * Nsrc + n] = v;
                }
            }
        }
    }
}

// ---------------------------------------------------------------------------
extern "C" void kernel_launch(void* const* d_in, const int* in_sizes, int n_in,
                              void* d_out, int out_size, void* d_ws, size_t ws_size,
                              hipStream_t stream)
{
    const long B = 32768;
    static const int K[5] = {1024, 1000, 800, 400, 600};
    static const int N[5] = {1000, 800, 400, 600, 6};

    int Kp[5], Np[5];
    for (int i = 0; i < 5; ++i) {
        Kp[i] = (K[i] + 31) & ~31;      // 1024,1024,800,416,608
        Np[i] = (N[i] + 127) & ~127;    // 1024,896,512,640,128
    }

    // workspace: padded f16 weights + padded f32 bias per layer (~5.5 MB)
    char* ws = (char*)d_ws;
    size_t off = 0;
    _Float16* Wp[5]; float* bp[5];
    for (int i = 0; i < 5; ++i) {
        Wp[i] = (_Float16*)(ws + off);
        off += (size_t)Np[i] * Kp[i] * sizeof(_Float16);
        off = (off + 127) & ~(size_t)127;
    }
    for (int i = 0; i < 5; ++i) {
        bp[i] = (float*)(ws + off);
        off += (size_t)Np[i] * sizeof(float);
        off = (off + 127) & ~(size_t)127;
    }

    // d_out layout: (h5, h1, h2, h3, h4, h5)
    float* outp = (float*)d_out;
    float* h5a = outp;
    float* h1  = h5a + B * 6;
    float* h2  = h1 + B * 1000;
    float* h3  = h2 + B * 800;
    float* h4  = h3 + B * 400;
    float* h5b = h4 + B * 600;

    // --- prep masked/padded weights ---
    for (int i = 0; i < 5; ++i) {
        const float* W = (const float*)d_in[1 + 3 * i];
        const float* b = (const float*)d_in[2 + 3 * i];
        const unsigned char* m = (const unsigned char*)d_in[3 + 3 * i];
        long total = (long)Np[i] * Kp[i];
        int blocks = (int)((total + 255) / 256);
        if (blocks > 8192) blocks = 8192;
        prep_weights<<<blocks, 256, 0, stream>>>(W, m, b, Wp[i], bp[i],
                                                 N[i], K[i], Np[i], Kp[i]);
    }

    // --- chained GEMMs, intermediates live in their d_out slots ---
    const float* ins[5]  = {(const float*)d_in[0], h1, h2, h3, h4};
    float*       outs[5] = {h1, h2, h3, h4, h5a};
    float*       out2[5] = {nullptr, nullptr, nullptr, nullptr, h5b};
    const int    relu[5] = {1, 1, 1, 0, 0};

    for (int i = 0; i < 5; ++i) {
        dim3 grid((unsigned)(B / 128), (unsigned)(Np[i] / 128));
        gemm_wmma_f16<<<grid, 256, 0, stream>>>(
            ins[i], K[i], Kp[i], Wp[i], bp[i],
            outs[i], out2[i], N[i], relu[i]);
    }
}
